// LSTMnet_85985245266218
// MI455X (gfx1250) — compile-verified, matrix operand order checked by reading, not run on hardware
//
#include <hip/hip_runtime.h>
#include <hip/hip_bf16.h>

// ---------------------------------------------------------------------------
// CDNA5 (gfx1250) LSTM: bf16 WMMA (v_wmma_f32_16x16x32_bf16) + TDM staging.
//   1. pack weights -> WMMA B-fragment order (bf16); GEMM weights grouped so
//      each block/kt weight slab is contiguous (one 16KB TDM transfer)
//   2. pad/convert inputs -> bf16 (K 130 -> 160)
//   3. GEMM x_proj0 = X * W_ih0^T + bias     [WMMA, TDM double-buffered B]
//   4. persistent recurrence layer0          [WMMA, h in LDS, c in VGPRs]
//   5. GEMM x_proj1, 6. recurrence layer1, 7. GEMM fc1 (+relu)
//   8. fc2 dot + group mean
// ---------------------------------------------------------------------------

typedef __attribute__((ext_vector_type(16))) __bf16 v16bf;
typedef __attribute__((ext_vector_type(8)))  __bf16 v8bf;
typedef __attribute__((ext_vector_type(8)))  float  v8f;

union V16U { v16bf v; v8bf h[2]; };

__device__ __forceinline__ v16bf combine16(v8bf lo, v8bf hi) {
    V16U u; u.h[0] = lo; u.h[1] = hi; return u.v;
}
__device__ __forceinline__ float sigmoidf_(float x) { return 1.0f / (1.0f + __expf(-x)); }
__device__ __forceinline__ float tanhf_(float x) {
    float e = __expf(2.0f * x); return (e - 1.0f) / (e + 1.0f);
}

// ---------------------------------------------------------------------------
// Tensor Data Mover: 1-D LDS fill (D# per CDNA5 ISA 08_async_tensor §8.3-8.4).
// ---------------------------------------------------------------------------
#if __has_builtin(__builtin_amdgcn_tensor_load_to_lds)
#define HAVE_TDM 1
#else
#define HAVE_TDM 0
#endif

#if HAVE_TDM
typedef unsigned int u32x4 __attribute__((ext_vector_type(4)));
typedef int          i32x4 __attribute__((ext_vector_type(4)));
typedef int          i32x8 __attribute__((ext_vector_type(8)));

__device__ __forceinline__ void tdm_load_1d(unsigned lds_addr,
                                            unsigned long long gaddr,
                                            unsigned ndwords) {
    // group0: count=1 (user desc), lds_addr, global_addr[56:0], type=2
    u32x4 g0 = { 1u,
                 lds_addr,
                 (unsigned)(gaddr & 0xFFFFFFFFull),
                 (unsigned)((gaddr >> 32) & 0x01FFFFFFull) | (2u << 30) };
    // group1: data_size=2 (4B); tensor_dim0 = tile_dim0 = ndwords (1-D tile,
    // tile_dim1=0 => higher dims unused); stride0 = ndwords (benign for 1-D).
    i32x8 g1 = { (int)(2u << 16),
                 (int)((ndwords & 0xFFFFu) << 16),   // tensor_dim0[15:0] @bits63:48
                 (int)((ndwords >> 16) & 0xFFFFu),   // tensor_dim0[31:16] @bits79:64
                 (int)((ndwords & 0xFFFFu) << 16),   // tile_dim0 @bits127:112
                 0,                                  // tile_dim1=0, tile_dim2=0
                 (int)ndwords,                       // tensor_dim0_stride lo
                 0, 0 };
    i32x4 gz = { 0, 0, 0, 0 };
#if defined(__clang_major__) && (__clang_major__ >= 23)
    i32x8 gz8 = { 0, 0, 0, 0, 0, 0, 0, 0 };
    __builtin_amdgcn_tensor_load_to_lds(g0, g1, gz, gz, gz8, 0);
#else
    __builtin_amdgcn_tensor_load_to_lds(g0, g1, gz, gz, 0);
#endif
}
__device__ __forceinline__ unsigned lds_off(const void* p) {
    return (unsigned)(unsigned long long)(__UINTPTR_TYPE__)p;
}
#endif

// ---------------------------------------------------------------------------
// Pack f32 weight W[N][K] (row-major) into WMMA B-fragment order (bf16).
//   n = ntile*16 + (lane&15)
//   k = kt*32 + koff, koff per ISA 16-bit operand layout:
//     lane<16 : j<8 ? j   : j+8     ({0..7,16..23})
//     lane>=16: j<8 ? j+8 : j+16    ({8..15,24..31})
// mode 0 (recurrence): idx = ((ntile*KT + kt)*32 + lane)*16 + j
// mode 1 (GEMM/TDM):   idx = (((g16*KT + kt)*16 + ntl)*32 + lane)*16 + j
//   with ntile = g16*16 + ntl  -> per-(block,kt) slab is contiguous 16KB.
// ---------------------------------------------------------------------------
__global__ void pack_wB(const float* __restrict__ W, __bf16* __restrict__ out,
                        int K, int KT, int mode, long total) {
    long idx = (long)blockIdx.x * blockDim.x + threadIdx.x;
    if (idx >= total) return;
    int  j    = (int)(idx & 15);
    int  lane = (int)((idx >> 4) & 31);
    long rest = idx >> 9;
    int  kt, ntile;
    if (mode == 0) {                 // ntile-major
        kt    = (int)(rest % KT);
        ntile = (int)(rest / KT);
    } else {                         // grouped-16, kt-major inside group
        int  ntl   = (int)(rest % 16);
        long rest2 = rest / 16;
        kt         = (int)(rest2 % KT);
        ntile      = (int)(rest2 / KT) * 16 + ntl;
    }
    int n = ntile * 16 + (lane & 15);
    int koff = (lane < 16) ? ((j < 8) ? j : j + 8)
                           : ((j < 8) ? j + 8 : j + 16);
    int k = kt * 32 + koff;
    float v = (k < K) ? W[(long)n * K + k] : 0.0f;
    out[idx] = (__bf16)v;
}

// f32 [R][K] -> bf16 [R][Kp] zero-padded
__global__ void cvt_pad_bf16(const float* __restrict__ in, __bf16* __restrict__ out,
                             int K, int Kp, long total) {
    long idx = (long)blockIdx.x * blockDim.x + threadIdx.x;
    if (idx >= total) return;
    long r = idx / Kp;
    int  k = (int)(idx % Kp);
    out[idx] = (k < K) ? (__bf16)in[r * (long)K + k] : (__bf16)0.0f;
}

// ---------------------------------------------------------------------------
// WMMA GEMM: C[M][crow] = A[M][Kp](bf16, row stride arow) * B + bias (+relu).
// Block = 256 thr = 8 waves (2 M x 4 N); tile 32 x 256.
// B weights: TDM DMA into double-buffered LDS slab (16KB per kt), consumed
// via ds_load_b128; wave0 runs the TENSORcnt producer protocol. Fallback:
// direct global b128 loads. A-fragments: two 16B global loads per lane,
// issued before the slab barriers so they overlap the TDM wait.
// ---------------------------------------------------------------------------
__global__ void __launch_bounds__(256) wmma_gemm_bf16(
    const __bf16* __restrict__ A, long arow,
    const __bf16* __restrict__ Bp, int KT,
    const float* __restrict__ bias0, const float* __restrict__ bias1,
    float* __restrict__ C, long crow, int relu)
{
    const int tid  = threadIdx.x;
    const int lane = tid & 31;
    const int w    = tid >> 5;
    const int wm   = w >> 2;        // 0..1
    const int wn   = w & 3;         // 0..3
    const long mbase = (long)blockIdx.y * 32 + wm * 16;
    const int  arowi = lane & 15;
    const int  aoff0 = (lane >= 16) ? 8 : 0;

    v8f acc[4];
    #pragma unroll
    for (int t = 0; t < 4; ++t)
        #pragma unroll
        for (int v = 0; v < 8; ++v) acc[t][v] = 0.0f;

#if HAVE_TDM
    __shared__ __align__(16) __bf16 bslab[2][16 * 32 * 16];   // 2 x 16KB
    const unsigned long long slab0 =
        (unsigned long long)(__UINTPTR_TYPE__)Bp + (unsigned long long)blockIdx.x * KT * 16384ull;
    if (w == 0) tdm_load_1d(lds_off(&bslab[0][0]), slab0, 4096);
    for (int kt = 0; kt < KT; ++kt) {
        const __bf16* ap = A + (mbase + arowi) * arow + kt * 32 + aoff0;
        v8bf alo = *(const v8bf*)ap;            // overlaps the TDM wait
        v8bf ahi = *(const v8bf*)(ap + 16);
        __syncthreads();                         // prior slab readers done
        if (w == 0) {
            if (kt + 1 < KT) {
                tdm_load_1d(lds_off(&bslab[(kt + 1) & 1][0]),
                            slab0 + (unsigned long long)(kt + 1) * 16384ull, 4096);
                __builtin_amdgcn_s_wait_tensorcnt(1);   // slab[kt] landed
            } else {
                __builtin_amdgcn_s_wait_tensorcnt(0);   // last slab landed
            }
        }
        __syncthreads();                         // slab[kt&1] visible to all
        v16bf a = combine16(alo, ahi);
        const v16bf* bl = (const v16bf*)&bslab[kt & 1][0];
        #pragma unroll
        for (int t = 0; t < 4; ++t) {
            v16bf b = bl[(wn * 4 + t) * 32 + lane];      // ds_load_b128 x2
            acc[t] = __builtin_amdgcn_wmma_f32_16x16x32_bf16(
                         false, a, false, b, (short)0, acc[t], false, false);
        }
    }
#else
    const v16bf* Bv = (const v16bf*)Bp;
    for (int kt = 0; kt < KT; ++kt) {
        const __bf16* ap = A + (mbase + arowi) * arow + kt * 32 + aoff0;
        v16bf a = combine16(*(const v8bf*)ap, *(const v8bf*)(ap + 16));
        if (kt + 1 < KT)
            __builtin_prefetch(
                (const void*)&Bv[((long)blockIdx.x * KT + kt + 1) * 16 * 32 + lane], 0, 3);
        #pragma unroll
        for (int t = 0; t < 4; ++t) {
            v16bf b = Bv[(((long)blockIdx.x * KT + kt) * 16 + wn * 4 + t) * 32 + lane];
            acc[t] = __builtin_amdgcn_wmma_f32_16x16x32_bf16(
                         false, a, false, b, (short)0, acc[t], false, false);
        }
    }
#endif

    #pragma unroll
    for (int t = 0; t < 4; ++t) {
        int n = (blockIdx.x * 16 + wn * 4 + t) * 16 + (lane & 15);
        float bv = 0.0f;
        if (bias0) bv += bias0[n];
        if (bias1) bv += bias1[n];
        long r0 = mbase + ((lane >= 16) ? 8 : 0);
        #pragma unroll
        for (int v = 0; v < 8; ++v) {
            float val = acc[t][v] + bv;
            if (relu) val = fmaxf(val, 0.0f);
            C[(r0 + v) * crow + n] = val;
        }
    }
}

// ---------------------------------------------------------------------------
// Persistent LSTM recurrence: one block owns 16 batch rows for all T steps.
// 512 thr = 16 waves; wave w owns h-columns [w*32,w*32+32) for ALL 4 gates
// so i/f/g/o combine without cross-wave traffic. h_{t-1} in LDS (row-major
// bf16, padded rows -> A-frags are two ds_load_b128), c in VGPRs, W_hh
// streamed from L2 (2MB bf16, resident in the 192MB L2 across 44 blocks).
// ---------------------------------------------------------------------------
__global__ void __launch_bounds__(512) lstm_recurrence(
    const float* __restrict__ xproj,   // [B*T][2048] f32 (pre-biased)
    const __bf16* __restrict__ Bp,     // packed W_hh fragments, NT=128, KT=16
    __bf16* __restrict__ hout,         // [B*T][512] bf16
    int T)
{
    __shared__ __align__(16) __bf16 hbuf[16][544];   // 512 + 32 pad (banking)
    const int  tid   = threadIdx.x;
    const int  lane  = tid & 31;
    const int  w     = tid >> 5;                 // 0..15
    const long b0    = (long)blockIdx.x * 16;
    const int  nloc  = lane & 15;
    const int  mhalf = (lane >= 16) ? 8 : 0;
    const int  aoff0 = (lane >= 16) ? 8 : 0;

    for (int i = tid; i < 16 * 544; i += blockDim.x)
        (&hbuf[0][0])[i] = (__bf16)0.0f;

    v8f acc[4][2];
    v8f c[2];
    #pragma unroll
    for (int nt = 0; nt < 2; ++nt)
        #pragma unroll
        for (int v = 0; v < 8; ++v) c[nt][v] = 0.0f;

    const v16bf* Bv = (const v16bf*)Bp;
    __syncthreads();

    for (int t = 0; t < T; ++t) {
        // gates := x_proj slice (bias already folded in)
        #pragma unroll
        for (int g = 0; g < 4; ++g)
            #pragma unroll
            for (int nt = 0; nt < 2; ++nt) {
                int col = g * 512 + w * 32 + nt * 16 + nloc;
                #pragma unroll
                for (int v = 0; v < 8; ++v) {
                    long r = (b0 + mhalf + v) * T + t;
                    acc[g][nt][v] = xproj[r * 2048 + col];
                }
            }
        // gates += h_{t-1} @ W_hh^T   (WMMA over K=512)
        for (int kt = 0; kt < 16; ++kt) {
            const __bf16* ap = &hbuf[nloc][kt * 32 + aoff0];
            v16bf a = combine16(*(const v8bf*)ap, *(const v8bf*)(ap + 16));
            if (kt + 1 < 16)
                __builtin_prefetch((const void*)&Bv[((long)(w * 2) * 16 + kt + 1) * 32 + lane], 0, 3);
            #pragma unroll
            for (int g = 0; g < 4; ++g)
                #pragma unroll
                for (int nt = 0; nt < 2; ++nt) {
                    int ntile = g * 32 + w * 2 + nt;
                    v16bf b = Bv[((long)ntile * 16 + kt) * 32 + lane];
                    acc[g][nt] = __builtin_amdgcn_wmma_f32_16x16x32_bf16(
                                     false, a, false, b, (short)0, acc[g][nt], false, false);
                }
        }
        // activations + state update; h parked in acc[0]
        #pragma unroll
        for (int nt = 0; nt < 2; ++nt) {
            int col = w * 32 + nt * 16 + nloc;
            #pragma unroll
            for (int v = 0; v < 8; ++v) {
                float ig = sigmoidf_(acc[0][nt][v]);
                float fg = sigmoidf_(acc[1][nt][v]);
                float gg = tanhf_   (acc[2][nt][v]);
                float og = sigmoidf_(acc[3][nt][v]);
                float cc = fg * c[nt][v] + ig * gg;
                c[nt][v] = cc;
                float h  = og * tanhf_(cc);
                acc[0][nt][v] = h;
                long r = (b0 + mhalf + v) * T + t;
                hout[r * 512 + col] = (__bf16)h;
            }
        }
        __syncthreads();   // all hbuf reads done before overwrite
        #pragma unroll
        for (int nt = 0; nt < 2; ++nt) {
            int col = w * 32 + nt * 16 + nloc;
            #pragma unroll
            for (int v = 0; v < 8; ++v)
                hbuf[mhalf + v][col] = (__bf16)acc[0][nt][v];
        }
        __syncthreads();   // h_t visible to every wave
    }
}

// fc2: per-row dot product (K=512) with wave32 shuffle reduction
__global__ void __launch_bounds__(256) fc2_reduce(
    const float* __restrict__ X, const float* __restrict__ W,
    const float* __restrict__ b, float* __restrict__ out, int M, int K)
{
    int lane = threadIdx.x & 31;
    int w    = threadIdx.x >> 5;
    int row  = blockIdx.x * 8 + w;
    if (row >= M) return;
    float s = 0.0f;
    for (int k = lane; k < K; k += 32) s += X[(long)row * K + k] * W[k];
    #pragma unroll
    for (int off = 16; off; off >>= 1) s += __shfl_xor(s, off, 32);
    if (lane == 0) out[row] = s + b[0];
}

__global__ void group_mean(const float* __restrict__ in, float* __restrict__ out,
                           int G, int GS) {
    int i = blockIdx.x * blockDim.x + threadIdx.x;
    if (i >= G) return;
    float s = 0.0f;
    for (int j = 0; j < GS; ++j) s += in[i * GS + j];
    out[i] = s / (float)GS;
}

// ---------------------------------------------------------------------------
extern "C" void kernel_launch(void* const* d_in, const int* in_sizes, int n_in,
                              void* d_out, int out_size, void* d_ws, size_t ws_size,
                              hipStream_t stream) {
    (void)in_sizes; (void)n_in; (void)out_size; (void)ws_size;
    const float* inputs = (const float*)d_in[0];
    const float* W_ih0  = (const float*)d_in[1];
    const float* W_hh0  = (const float*)d_in[2];
    const float* b_ih0  = (const float*)d_in[3];
    const float* b_hh0  = (const float*)d_in[4];
    const float* W_ih1  = (const float*)d_in[5];
    const float* W_hh1  = (const float*)d_in[6];
    const float* b_ih1  = (const float*)d_in[7];
    const float* b_hh1  = (const float*)d_in[8];
    const float* W_fc1  = (const float*)d_in[9];
    const float* b_fc1  = (const float*)d_in[10];
    const float* W_fc2  = (const float*)d_in[11];
    const float* b_fc2  = (const float*)d_in[12];
    float* out = (float*)d_out;

    const long B = 704, T = 128, H = 512, G4 = 2048;
    const long BT = B * T;                       // 90112
    const long Kp0 = 160;                        // 130 padded to 32-mult

    // workspace carve-up (all chunks 256B-multiples)
    char* p = (char*)d_ws;
    __bf16* pW_ih0 = (__bf16*)p;  p += G4 * Kp0 * 2;     // 640 KB
    __bf16* pW_hh0 = (__bf16*)p;  p += G4 * H   * 2;     // 2 MB
    __bf16* pW_ih1 = (__bf16*)p;  p += G4 * H   * 2;     // 2 MB
    __bf16* pW_hh1 = (__bf16*)p;  p += G4 * H   * 2;     // 2 MB
    __bf16* pW_fc1 = (__bf16*)p;  p += H  * H   * 2;     // 512 KB
    __bf16* Ain    = (__bf16*)p;  p += BT * Kp0 * 2;     // ~29 MB
    __bf16* h1     = (__bf16*)p;  p += BT * H   * 2;     // ~92 MB
    __bf16* h2     = (__bf16*)p;  p += BT * H   * 2;     // ~92 MB
    float*  fc1o   = (float*)p;   p += B  * H   * 4;     // ~1.4 MB
    float*  xproj  = (float*)p;                          // ~738 MB (reused L0/L1)

    // 1. pack weights: GEMM weights grouped-16 (mode 1, TDM slabs),
    //    recurrence weights ntile-major (mode 0)
    pack_wB<<<(unsigned)((G4 * Kp0) / 256), 256, 0, stream>>>(W_ih0, pW_ih0, 130, (int)(Kp0 / 32), 1, G4 * Kp0);
    pack_wB<<<(unsigned)((G4 * H) / 256),   256, 0, stream>>>(W_hh0, pW_hh0, (int)H, 16, 0, G4 * H);
    pack_wB<<<(unsigned)((G4 * H) / 256),   256, 0, stream>>>(W_ih1, pW_ih1, (int)H, 16, 1, G4 * H);
    pack_wB<<<(unsigned)((G4 * H) / 256),   256, 0, stream>>>(W_hh1, pW_hh1, (int)H, 16, 0, G4 * H);
    pack_wB<<<(unsigned)((H * H) / 256),    256, 0, stream>>>(W_fc1, pW_fc1, (int)H, 16, 1, H * H);

    // 2. inputs f32[BT][130] -> bf16[BT][160]
    cvt_pad_bf16<<<(unsigned)((BT * Kp0) / 256), 256, 0, stream>>>(inputs, Ain, 130, (int)Kp0, BT * Kp0);

    // 3. x_proj0 (+ biases)
    wmma_gemm_bf16<<<dim3(8, (unsigned)(BT / 32)), 256, 0, stream>>>(
        Ain, Kp0, pW_ih0, (int)(Kp0 / 32), b_ih0, b_hh0, xproj, G4, 0);

    // 4. layer-0 recurrence
    lstm_recurrence<<<(unsigned)(B / 16), 512, 0, stream>>>(xproj, pW_hh0, h1, (int)T);

    // 5. x_proj1 (+ biases)
    wmma_gemm_bf16<<<dim3(8, (unsigned)(BT / 32)), 256, 0, stream>>>(
        h1, H, pW_ih1, 16, b_ih1, b_hh1, xproj, G4, 0);

    // 6. layer-1 recurrence
    lstm_recurrence<<<(unsigned)(B / 16), 512, 0, stream>>>(xproj, pW_hh1, h2, (int)T);

    // 7. fc1 = relu(last @ W_fc1^T + b): A = h2 rows at t = T-1, stride T*H
    wmma_gemm_bf16<<<dim3(2, (unsigned)(B / 32)), 256, 0, stream>>>(
        h2 + (T - 1) * H, T * H, pW_fc1, 16, b_fc1, nullptr, fc1o, H, 1);

    // 8. fc2 + group mean
    fc2_reduce<<<(unsigned)(B / 8), 256, 0, stream>>>(fc1o, W_fc2, b_fc2, out, (int)B, (int)H);
    group_mean<<<1, 32, 0, stream>>>(out, out + B, 32, 22);
}
